// DPM_39101382263216
// MI455X (gfx1250) — matrix operand order repaired and checked by reading, not compiled, not run on hardware
//
#include <hip/hip_runtime.h>

typedef __attribute__((ext_vector_type(16))) _Float16 v16h;
typedef __attribute__((ext_vector_type(8)))  float    v8f;

#define S_TOT   49152          // B*T = 4096*12
#define N_NODES 50
#define HIDV    6
#define INSZ    12
#define MROWS   (S_TOT*HIDV)   // 294912 im2col rows for the convs
#define MTILES  (MROWS/16)     // 18432 (divides exactly)

__device__ __forceinline__ float fsigmoid(float x) {
    return 1.0f / (1.0f + __expf(-x));
}
__device__ __forceinline__ float ftanh_fast(float x) {
    x = fminf(fmaxf(x, -15.0f), 15.0f);
    float e = __expf(2.0f * x);
    return (e - 1.0f) / (e + 1.0f);
}

// ---------------------------------------------------------------------------
// Serial LSTM scan: 50 independent node-chains, one wave32 per node.
// Lane g (<24) owns gate g; weights live in registers; h/c replicated across
// lanes via shuffles. x-projection folded in (off the h-critical path),
// next-step x prefetched. Output h written as f16 in [S][node][hid] layout.
// ---------------------------------------------------------------------------
__global__ void __launch_bounds__(32)
lstm_scan_kernel(const float* __restrict__ X,
                 const float* __restrict__ W_ih, const float* __restrict__ W_hh,
                 const float* __restrict__ b_ih, const float* __restrict__ b_hh,
                 _Float16* __restrict__ al)
{
    const int node = blockIdx.x;
    const int lane = threadIdx.x;
    const int g    = (lane < 24) ? lane : 0;

    float wih[INSZ], whh[HIDV];
#pragma unroll
    for (int k = 0; k < INSZ; ++k) wih[k] = W_ih[g*INSZ + k];
#pragma unroll
    for (int j = 0; j < HIDV; ++j) whh[j] = W_hh[g*HIDV + j];
    const float bsum = b_ih[g] + b_hh[g];
    const bool  gate_is_tanh = (lane >= 12 && lane < 18);   // "g" gate block

    float h[HIDV], c[HIDV];
#pragma unroll
    for (int j = 0; j < HIDV; ++j) { h[j] = 0.0f; c[j] = 0.0f; }

    float xv = (lane < INSZ) ? X[(0*N_NODES + node)*INSZ + lane] : 0.0f;

    for (int t = 0; t < S_TOT; ++t) {
        float xv_next = 0.0f;
        if (t + 1 < S_TOT && lane < INSZ)
            xv_next = X[((size_t)(t+1)*N_NODES + node)*INSZ + lane];

        float gate = bsum;
#pragma unroll
        for (int k = 0; k < INSZ; ++k)
            gate += __shfl(xv, k, 32) * wih[k];
#pragma unroll
        for (int j = 0; j < HIDV; ++j)
            gate += h[j] * whh[j];

        float a = gate_is_tanh ? ftanh_fast(gate) : fsigmoid(gate);

#pragma unroll
        for (int j = 0; j < HIDV; ++j) {
            float iv = __shfl(a, j,      32);
            float fv = __shfl(a, j + 6,  32);
            float gv = __shfl(a, j + 12, 32);
            float ov = __shfl(a, j + 18, 32);
            c[j] = fv * c[j] + iv * gv;
            h[j] = ftanh_fast(ov * ftanh_fast(c[j]));   // extra tanh per ref
        }
        if (lane < HIDV)
            al[((size_t)t*N_NODES + node)*HIDV + lane] = (_Float16)h[lane];
        xv = xv_next;
    }
}

// ---------------------------------------------------------------------------
// 3x3 conv on [S][Cin][6][1] (pad 1) == 1-D k=3 conv along H=6, as an im2col
// GEMM: rows=(s,h), K=Cin*3, N=Cout. One wave per 16-row M-tile, WMMA
// f32_16x16x32_f16, weights pre-packed into LDS in B-fragment lane order.
// ---------------------------------------------------------------------------
template<int CIN, int COUT, int KC, int NT>
__global__ void __launch_bounds__(256)
conv_wmma_kernel(const _Float16* __restrict__ in,
                 const float* __restrict__ W,     // [COUT][CIN][3][3]
                 const float* __restrict__ bias,
                 _Float16* __restrict__ out)
{
    extern __shared__ char smem_raw[];
    _Float16* lds_b   = (_Float16*)smem_raw;                       // KC*NT*512 halves
    float*    lds_bias= (float*)(smem_raw + (size_t)KC*NT*512*sizeof(_Float16));

    const int tid = threadIdx.x;
    const int KW  = 3 * CIN;

    // Stage weights directly in B-fragment order: lane L holds K=(L>=16?16:0)+j.
    for (int idx = tid; idx < KC*NT*512; idx += blockDim.x) {
        int j    =  idx        & 15;
        int lane = (idx >> 4)  & 31;
        int nt   = (idx >> 9)  % NT;
        int kc   =  idx / (512 * NT);
        int k    = kc*32 + ((lane & 16) ? 16 : 0) + j;
        int n    = nt*16 + (lane & 15);
        _Float16 w = (_Float16)0.0f;
        if (k < KW && n < COUT) {
            int cch = k / 3, tt = k % 3;
            w = (_Float16)W[((n*CIN + cch)*3 + tt)*3 + 1];  // only mid column hits data
        }
        lds_b[idx] = w;
    }
    for (int idx = tid; idx < COUT; idx += blockDim.x)
        lds_bias[idx] = bias[idx];
    __syncthreads();

    const int wave = tid >> 5;
    const int lane = tid & 31;
    const int m    = lane & 15;
    const int hi   = (lane & 16) ? 1 : 0;

    union { v16h v; _Float16 e[16]; } afrag;
    union { v8f  v; float    e[8];  } acc[NT];

    for (int mt = blockIdx.x*8 + wave; mt < MTILES; mt += gridDim.x*8) {
#pragma unroll
        for (int nt = 0; nt < NT; ++nt)
#pragma unroll
            for (int r = 0; r < 8; ++r) acc[nt].e[r] = 0.0f;

        const int row = mt*16 + m;
        const int s   = row / 6;
        const int hh0 = row % 6;

        for (int kc = 0; kc < KC; ++kc) {
            // A fragment: K = j + 16*(lane>=16) (matches B staging)
#pragma unroll
            for (int j = 0; j < 16; ++j) {
                int kl = j + (hi ? 16 : 0);
                int k  = kc*32 + kl;
                _Float16 av = (_Float16)0.0f;
                if (k < KW) {
                    int cch = k / 3, tt = k % 3;
                    int hh  = hh0 + tt - 1;
                    if (hh >= 0 && hh < 6)
                        av = in[((size_t)s*CIN + cch)*6 + hh];
                }
                afrag.e[j] = av;
            }
#pragma unroll
            for (int nt = 0; nt < NT; ++nt) {
                v16h bfrag = *(const v16h*)&lds_b[((size_t)(kc*NT + nt)*32 + lane)*16];
                acc[nt].v = __builtin_amdgcn_wmma_f32_16x16x32_f16(
                    false, afrag.v, false, bfrag, (short)0, acc[nt].v, false, false);
            }
        }

        // Epilogue: D row = r + 8*(lane>=16), col = lane%16; bias + ReLU, f16 store
#pragma unroll
        for (int nt = 0; nt < NT; ++nt) {
            int o = nt*16 + (lane & 15);
            if (o >= COUT) continue;
            float bv = lds_bias[o];
#pragma unroll
            for (int r = 0; r < 8; ++r) {
                int orow = mt*16 + r + (hi ? 8 : 0);
                int s2 = orow / 6, h2 = orow % 6;
                float val = fmaxf(acc[nt].e[r] + bv, 0.0f);
                out[((size_t)s2*COUT + o)*6 + h2] = (_Float16)val;
            }
        }
    }
}

// ---------------------------------------------------------------------------
// Final 6x6 "lean" layer: one thread per (s, node) row.
// ---------------------------------------------------------------------------
__global__ void __launch_bounds__(256)
lean_kernel(const _Float16* __restrict__ f3,
            const float* __restrict__ W_lean, const float* __restrict__ b_lean,
            float* __restrict__ outp)
{
    int idx = blockIdx.x * blockDim.x + threadIdx.x;     // = s*50 + node
    if (idx >= S_TOT * N_NODES) return;
    float fv[HIDV];
#pragma unroll
    for (int hh = 0; hh < HIDV; ++hh) fv[hh] = (float)f3[(size_t)idx*HIDV + hh];
#pragma unroll
    for (int j = 0; j < HIDV; ++j) {
        float a = b_lean[j];
#pragma unroll
        for (int hh = 0; hh < HIDV; ++hh) a += fv[hh] * W_lean[j*HIDV + hh];
        outp[(size_t)idx*HIDV + j] = a;
    }
}

extern "C" void kernel_launch(void* const* d_in, const int* in_sizes, int n_in,
                              void* d_out, int out_size, void* d_ws, size_t ws_size,
                              hipStream_t stream)
{
    // setup_inputs order: A_hat(0, unused!), X(1), W_ih(2), W_hh(3), b_ih(4),
    // b_hh(5), Wf1(6), bf1(7), Wf2(8), bf2(9), Wf3(10), bf3(11), W_lean(12), b_lean(13)
    const float* X      = (const float*)d_in[1];
    const float* W_ih   = (const float*)d_in[2];
    const float* W_hh   = (const float*)d_in[3];
    const float* b_ih   = (const float*)d_in[4];
    const float* b_hh   = (const float*)d_in[5];
    const float* Wf1    = (const float*)d_in[6];
    const float* bf1    = (const float*)d_in[7];
    const float* Wf2    = (const float*)d_in[8];
    const float* bf2    = (const float*)d_in[9];
    const float* Wf3    = (const float*)d_in[10];
    const float* bf3    = (const float*)d_in[11];
    const float* W_lean = (const float*)d_in[12];
    const float* b_lean = (const float*)d_in[13];
    float* outp = (float*)d_out;

    // f16 workspace: al | f1 | f2 ; f3 reuses al's region (al dead after conv1)
    _Float16* al = (_Float16*)d_ws;
    _Float16* f1 = al + (size_t)S_TOT * N_NODES * HIDV;   // 14.7M halves
    _Float16* f2 = f1 + (size_t)S_TOT * 100 * HIDV;       // 29.5M halves
    _Float16* f3 = al;

    lstm_scan_kernel<<<N_NODES, 32, 0, stream>>>(X, W_ih, W_hh, b_ih, b_hh, al);

    size_t sm1 = (size_t)5  * 7 * 512 * sizeof(_Float16) + 100 * sizeof(float);
    conv_wmma_kernel<50, 100, 5, 7><<<288, 256, sm1, stream>>>(al, Wf1, bf1, f1);

    size_t sm2 = (size_t)10 * 7 * 512 * sizeof(_Float16) + 100 * sizeof(float);
    conv_wmma_kernel<100, 100, 10, 7><<<288, 256, sm2, stream>>>(f1, Wf2, bf2, f2);

    size_t sm3 = (size_t)10 * 4 * 512 * sizeof(_Float16) + 50 * sizeof(float);
    conv_wmma_kernel<100, 50, 10, 4><<<288, 256, sm3, stream>>>(f2, Wf3, bf3, f3);

    lean_kernel<<<(S_TOT*N_NODES + 255)/256, 256, 0, stream>>>(f3, W_lean, b_lean, outp);
}
